// Model_61065845014888
// MI455X (gfx1250) — matrix operand order repaired and checked by reading, not compiled, not run on hardware
//
#include <hip/hip_runtime.h>

typedef _Float16 v16h __attribute__((ext_vector_type(16)));
typedef _Float16 v8h  __attribute__((ext_vector_type(8)));
typedef float    v8f  __attribute__((ext_vector_type(8)));

#define NPTS       2097152
#define LEVELS     16
#define GRID_P     513                     // R+1
#define LVL_STRIDE (513*513)               // float2 elements per level
#define HID        128
#define W1T_STRIDE 40                      // halfs: 32 + 8 pad  (row advance = 4 dwords -> conflict free)
#define W2T_STRIDE 136                     // halfs: 128 + 8 pad
#define W2T_OFF    (128*W1T_STRIDE)        // 5120 halfs
#define SMW_HALFS  (W2T_OFF + 128*W2T_STRIDE) // 22528 halfs = 45056 B
#define ACT_STRIDE 136                     // halfs per activation row

__device__ constexpr int kRes[LEVELS] = {
    16, 20, 25, 32, 40, 50, 64, 80, 101, 128, 161, 203, 256, 322, 406, 512
};

// ---------------------------------------------------------------------------
// Prep: convert W1 (32x128) and W2 (128x128) to transposed f16 images in d_ws,
// laid out exactly as the main kernel's LDS wants them (padded row strides).
// B-fragment for WMMA then becomes two runs of 8 contiguous halfs per lane.
// ---------------------------------------------------------------------------
__global__ void prep_weights(const float* __restrict__ W1,
                             const float* __restrict__ W2,
                             _Float16* __restrict__ ws)
{
    int tid = blockIdx.x * blockDim.x + threadIdx.x;   // 16384 threads
    if (tid < 32 * HID) {                              // W1: coalesced read, transposed write
        int i = tid >> 7, o = tid & 127;               // W1[i][o]
        ws[o * W1T_STRIDE + i] = (_Float16)W1[tid];
    }
    {                                                  // W2: 16384 elements
        int i = tid >> 7, o = tid & 127;               // W2[i][o]
        ws[W2T_OFF + o * W2T_STRIDE + i] = (_Float16)W2[tid];
    }
}

// ---------------------------------------------------------------------------
// Main fused kernel: hash-grid bilerp encode + 3-layer MLP via WMMA.
// Block = 128 threads = 4 waves; each wave owns 16 points (one M-tile).
// ---------------------------------------------------------------------------
__global__ __launch_bounds__(128) void hashgrid_mlp(
    const float2*  __restrict__ x,
    const float2*  __restrict__ grids,
    const float*   __restrict__ b1,
    const float*   __restrict__ b2,
    const float*   __restrict__ W3,
    const float*   __restrict__ b3,
    const _Float16* __restrict__ wsW,
    float*         __restrict__ out)
{
    __shared__ __align__(16) _Float16 smW[SMW_HALFS];           // 45056 B
    __shared__ __align__(16) _Float16 smAct[4][16][ACT_STRIDE]; // 17408 B

    // ---- cooperative flat copy of pre-transposed f16 weights into LDS ----
    {
        const uint4* src = (const uint4*)wsW;
        uint4*       dst = (uint4*)smW;
        #pragma unroll 4
        for (int i = threadIdx.x; i < SMW_HALFS / 8; i += 128)
            dst[i] = src[i];
    }
    __syncthreads();

    const int lane  = threadIdx.x & 31;
    const int wave  = threadIdx.x >> 5;
    const int m     = lane & 15;      // A-layout: lane%16 = M (point row)
    const int khalf = lane >> 4;      // lane half selects K sub-range
    const long gbase = ((long)blockIdx.x * 4 + wave) * 16;

    const float2 xy = x[gbase + m];

    // ------------------------------------------------------------------
    // Feature encode: each lane bilerps the 8 levels whose K-slots this
    // lane owns in the 16x32 f16 A-fragment layout:
    //   element e(0..7)  -> K = khalf*8 + e        (levels khalf*4 + 0..3, and +8)
    //   element e(8..15) -> K = 16 + khalf*8 + e-8
    // ------------------------------------------------------------------
    v16h a1;
    #pragma unroll
    for (int g = 0; g < 2; ++g) {
        #pragma unroll
        for (int j = 0; j < 4; ++j) {
            const int lvLo = g * 8 + j;        // level if khalf == 0
            const int lvHi = g * 8 + 4 + j;    // level if khalf == 1
            const int r    = khalf ? kRes[lvHi] : kRes[lvLo];
            const float2* gl = grids + (long)(khalf ? lvHi : lvLo) * LVL_STRIDE;

            const float rf = (float)r;
            const float px = xy.x * rf, py = xy.y * rf;
            const float fx = floorf(px), fy = floorf(py);
            const float wx = px - fx,   wy = py - fy;
            int i0x = (int)fx; i0x = i0x < 0 ? 0 : (i0x > r ? r : i0x);
            int i0y = (int)fy; i0y = i0y < 0 ? 0 : (i0y > r ? r : i0y);
            const int i1x = (i0x + 1 > r) ? r : i0x + 1;
            const int i1y = (i0y + 1 > r) ? r : i0y + 1;

            const float2 f00 = gl[i0x * GRID_P + i0y];
            const float2 f10 = gl[i1x * GRID_P + i0y];
            const float2 f01 = gl[i0x * GRID_P + i1y];
            const float2 f11 = gl[i1x * GRID_P + i1y];

            const float ax = 1.f - wx, ay = 1.f - wy;
            const float v0 = (f00.x * ax + f10.x * wx) * ay + (f01.x * ax + f11.x * wx) * wy;
            const float v1 = (f00.y * ax + f10.y * wx) * ay + (f01.y * ax + f11.y * wx) * wy;
            a1[g * 8 + 2 * j]     = (_Float16)v0;
            a1[g * 8 + 2 * j + 1] = (_Float16)v1;
        }
    }

    // ------------------------------------------------------------------
    // Layer 1: h1 = relu(feat(16x32) @ W1(32x128) + b1)  — 8 WMMAs, K=32.
    // D tile -> ReLU -> f16 -> LDS (transpose to A-layout happens via LDS).
    // ------------------------------------------------------------------
    #pragma unroll
    for (int t = 0; t < 8; ++t) {
        const _Float16* wr = smW + (t * 16 + m) * W1T_STRIDE + khalf * 8;
        const v8h blo = *(const v8h*)(wr);
        const v8h bhi = *(const v8h*)(wr + 16);
        const v16h b = __builtin_shufflevector(blo, bhi,
                       0,1,2,3,4,5,6,7,8,9,10,11,12,13,14,15);
        const float bias = b1[t * 16 + m];
        v8f c = {bias, bias, bias, bias, bias, bias, bias, bias};
        c = __builtin_amdgcn_wmma_f32_16x16x32_f16(
                false, a1, false, b, (short)0, c, false, false);
        #pragma unroll
        for (int v = 0; v < 8; ++v) {
            const float h = c[v] > 0.f ? c[v] : 0.f;    // ReLU
            // D element (vgpr v, lane) = row M = v + 8*khalf, col N = t*16+m
            smAct[wave][v + 8 * khalf][t * 16 + m] = (_Float16)h;
        }
    }
    // Intra-wave LDS RAW: DS ops from the same wave complete in order;
    // compiler inserts the s_wait_dscnt before dependent reads.

    // ---- reload h1 as four 16x32 A-fragments (reused across all 8 N-tiles) ----
    v16h a2[4];
    {
        const _Float16* ar = &smAct[wave][m][0];
        #pragma unroll
        for (int kt = 0; kt < 4; ++kt) {
            const v8h lo = *(const v8h*)(ar + kt * 32 + khalf * 8);
            const v8h hi = *(const v8h*)(ar + kt * 32 + 16 + khalf * 8);
            a2[kt] = __builtin_shufflevector(lo, hi,
                     0,1,2,3,4,5,6,7,8,9,10,11,12,13,14,15);
        }
    }

    // ------------------------------------------------------------------
    // Layer 2 (32 WMMAs) with layer 3 fused into VALU partials:
    //   out[m] = sum_n relu(h2[m][n]) * W3[n] + b3
    // ------------------------------------------------------------------
    float r3[8] = {0.f, 0.f, 0.f, 0.f, 0.f, 0.f, 0.f, 0.f};
    #pragma unroll
    for (int t = 0; t < 8; ++t) {
        const float bias = b2[t * 16 + m];
        v8f c = {bias, bias, bias, bias, bias, bias, bias, bias};
        #pragma unroll
        for (int kt = 0; kt < 4; ++kt) {
            const _Float16* wr = smW + W2T_OFF + (t * 16 + m) * W2T_STRIDE
                               + kt * 32 + khalf * 8;
            const v8h blo = *(const v8h*)(wr);
            const v8h bhi = *(const v8h*)(wr + 16);
            const v16h b = __builtin_shufflevector(blo, bhi,
                           0,1,2,3,4,5,6,7,8,9,10,11,12,13,14,15);
            c = __builtin_amdgcn_wmma_f32_16x16x32_f16(
                    false, a2[kt], false, b, (short)0, c, false, false);
        }
        const float w3v = W3[t * 16 + m];        // W3 value for column N = t*16+m
        #pragma unroll
        for (int v = 0; v < 8; ++v) {
            const float h = c[v] > 0.f ? c[v] : 0.f;   // ReLU on h2
            r3[v] = fmaf(h, w3v, r3[v]);
        }
    }

    // ---- reduce partials across the 16-lane N dimension (per lane-half) ----
    #pragma unroll
    for (int v = 0; v < 8; ++v) {
        float s = r3[v];
        s += __shfl_xor(s, 1, 32);
        s += __shfl_xor(s, 2, 32);
        s += __shfl_xor(s, 4, 32);
        s += __shfl_xor(s, 8, 32);
        r3[v] = s;
    }
    const float b3v = b3[0];
    #pragma unroll
    for (int v = 0; v < 8; ++v)
        if (m == v)                                   // one writer per output row
            out[gbase + khalf * 8 + v] = r3[v] + b3v;
}

// ---------------------------------------------------------------------------
extern "C" void kernel_launch(void* const* d_in, const int* in_sizes, int n_in,
                              void* d_out, int out_size, void* d_ws, size_t ws_size,
                              hipStream_t stream)
{
    const float* x     = (const float*)d_in[0];
    const float* grids = (const float*)d_in[1];
    const float* W1    = (const float*)d_in[2];
    const float* b1    = (const float*)d_in[3];
    const float* W2    = (const float*)d_in[4];
    const float* b2    = (const float*)d_in[5];
    const float* W3    = (const float*)d_in[6];
    const float* b3    = (const float*)d_in[7];
    _Float16* wsW = (_Float16*)d_ws;   // needs 45056 bytes

    // Stage 1: transpose + f16-convert weights into scratch (once per call).
    prep_weights<<<64, 256, 0, stream>>>(W1, W2, wsW);

    // Stage 2: fused encode + MLP. 64 points per 128-thread block.
    const int nblocks = NPTS / 64;     // 32768
    hashgrid_mlp<<<nblocks, 128, 0, stream>>>(
        (const float2*)x, (const float2*)grids,
        b1, b2, W3, b3, wsW, (float*)d_out);
}
